// GlocaTransformerBlock2D_747324309600
// MI455X (gfx1250) — matrix operand order repaired and checked by reading, not compiled, not run on hardware
//
#include <hip/hip_runtime.h>
#include <hip/hip_bf16.h>

typedef _Float16 h8  __attribute__((ext_vector_type(8)));
typedef _Float16 h4  __attribute__((ext_vector_type(4)));
typedef _Float16 v16h __attribute__((ext_vector_type(16)));
typedef float    v8f  __attribute__((ext_vector_type(8)));

#define CC    256
#define LL    1024
#define R1_   32
#define NHEAD 8
#define DH_   32
#define HD_   16384
#define MID_  1024

#define MBLK 128
#define NBLK 128
#define KC   32
#define LDSS 40   // halves; 80B row stride -> 16B aligned sub-reads

__device__ __forceinline__ v16h mk16(h8 lo, h8 hi) {
  v16h r;
#pragma unroll
  for (int j = 0; j < 8; ++j) { r[j] = lo[j]; r[j + 8] = hi[j]; }
  return r;
}

// Async DMA of 16 bytes global -> LDS (per lane), tracked by ASYNCcnt.
// LDS address = low 32 bits of the generic pointer.
__device__ __forceinline__ void async_b128(_Float16* lds_dst, const _Float16* g_src) {
  const unsigned l = (unsigned)(uintptr_t)lds_dst;
  asm volatile("global_load_async_to_lds_b128 %0, %1, off"
               :: "v"(l), "v"(g_src) : "memory");
}
__device__ __forceinline__ void wait_async0() {
  asm volatile("s_wait_asynccnt 0x0" ::: "memory");
}

// ---------------------------------------------------------------------------
// f32 -> f16 convert (weights are already (N,K) row-major == B^T; no transpose
// needed because the WMMA B-fragment is a contiguous read of a B^T row).
// ---------------------------------------------------------------------------
__global__ __launch_bounds__(256) void k_convert_f16(
    const float* __restrict__ src, _Float16* __restrict__ dst, long n) {
  long i4 = ((long)blockIdx.x * 256 + threadIdx.x) * 4;
  if (i4 < n) {
    float4 v = *(const float4*)(src + i4);
    h4 o = {(_Float16)v.x, (_Float16)v.y, (_Float16)v.z, (_Float16)v.w};
    *(h4*)(dst + i4) = o;
  }
}

// ---------------------------------------------------------------------------
// Build V^T:  vT[b][h][d][key] = qkv[b*L+key][512 + h*32 + d]
// ---------------------------------------------------------------------------
__global__ __launch_bounds__(256) void k_build_vT(
    const _Float16* __restrict__ qkvb, _Float16* __restrict__ vT) {
  const size_t idx = (size_t)blockIdx.x * 256 + threadIdx.x;  // 2*8*32*1024
  const int key = (int)(idx & 1023);
  const int d   = (int)((idx >> 10) & 31);
  const int h   = (int)((idx >> 15) & 7);
  const int b   = (int)(idx >> 18);
  vT[idx] = qkvb[((size_t)(b * LL + key)) * 768 + 512 + h * DH_ + d];
}

// ---------------------------------------------------------------------------
// LayerNorm over C=256, one wave per pixel; rearr=1 also does the unfold
// rearrange  (B,H,W,C) -> (B, L, P0*P1*C)  used by the down-projection.
// ---------------------------------------------------------------------------
__global__ __launch_bounds__(256) void k_ln(
    const float* __restrict__ in, const float* __restrict__ gamma,
    const float* __restrict__ beta, _Float16* __restrict__ outp, int rearr) {
  const int lane = threadIdx.x & 31;
  const int wave = threadIdx.x >> 5;
  const size_t pix = (size_t)blockIdx.x * 8 + wave;   // 0 .. B*H*W-1
  const float* xp = in + pix * CC + lane * 8;
  float4 v0 = *(const float4*)xp;
  float4 v1 = *(const float4*)(xp + 4);
  float v[8] = {v0.x, v0.y, v0.z, v0.w, v1.x, v1.y, v1.z, v1.w};
  float s = 0.f, s2 = 0.f;
#pragma unroll
  for (int i = 0; i < 8; ++i) { s += v[i]; s2 += v[i] * v[i]; }
#pragma unroll
  for (int m = 16; m >= 1; m >>= 1) {
    s  += __shfl_xor(s,  m, 32);
    s2 += __shfl_xor(s2, m, 32);
  }
  const float mean = s * (1.f / CC);
  const float rstd = rsqrtf(s2 * (1.f / CC) - mean * mean + 1e-5f);
  _Float16* op;
  if (rearr) {
    const int bidx = (int)(pix >> 16);
    const int hh = (int)((pix >> 8) & 255);
    const int ww = (int)(pix & 255);
    const int r0 = hh >> 3, p0 = hh & 7, r1 = ww >> 3, p1 = ww & 7;
    const size_t token = (size_t)bidx * LL + r0 * R1_ + r1;
    op = outp + token * HD_ + (size_t)((p0 << 3) + p1) * 256 + lane * 8;
  } else {
    op = outp + pix * CC + lane * 8;
  }
#pragma unroll
  for (int i = 0; i < 8; ++i) {
    const int c = lane * 8 + i;
    op[i] = (_Float16)((v[i] - mean) * rstd * gamma[c] + beta[c]);
  }
}

// ---------------------------------------------------------------------------
// Batched WMMA GEMM:  Out = A(MxK,f16) * Bt^T  (+bias, epilogue)
//   A tile: async global->LDS DMA, double-buffered, ONE barrier per K-step.
//   B read directly from global as (N,K) row-major -> B-fragment = two b128s.
// Block 256 thr = 8 waves; tile 128x128; wave tile 32x64 = 8 WMMA / K-step.
// M must be a multiple of 128. NGUARD=true clamps B columns (stores guarded).
// mode 0: f16 out (+bias)        mode 1: f16 out, bias + exact GELU
// mode 2: fold (B,L,HD)->(B,H,W,C), +bias +resid, f32 out
// mode 3: f32 out, +bias, in-place residual add (Out += ...)
// ---------------------------------------------------------------------------
template <int MODE, bool NGUARD>
__global__ __launch_bounds__(256) void k_gemm(
    const _Float16* __restrict__ A, int lda, long sAo, long sAi,
    const _Float16* __restrict__ Bt, long sBo, long sBi,
    void* __restrict__ Out, int ldo, long sOo, long sOi,
    const float* __restrict__ bias, const float* __restrict__ resid,
    int M, int N, int K, int zInner) {
  __shared__ _Float16 sA0[MBLK * LDSS];
  __shared__ _Float16 sA1[MBLK * LDSS];
  const int tid  = threadIdx.x;
  const int lane = tid & 31;
  const int wave = tid >> 5;
  const int wm = wave >> 1;              // 0..3  (M direction)
  const int wn = wave & 1;               // 0..1  (N direction)
  const int z  = blockIdx.z;
  const int zo = z / zInner, zi = z % zInner;
  const _Float16* Ag = A  + (size_t)zo * sAo + (size_t)zi * sAi;
  const _Float16* Bg = Bt + (size_t)zo * sBo + (size_t)zi * sBi;
  const size_t obase = (size_t)zo * sOo + (size_t)zi * sOi;
  const int m0 = blockIdx.y * MBLK;
  const int n0 = blockIdx.x * NBLK;
  const int r    = lane & 15;
  const int half = lane >> 4;

  // A staging: 128x32 halves; thread covers rows ar0 and ar0+64, 8 halves each
  const int ar0 = tid >> 2;
  const int ac0 = (tid & 3) << 3;
  const int so0 = ar0 * LDSS + ac0;
  const int so1 = (ar0 + 64) * LDSS + ac0;
  const _Float16* Arow0 = Ag + (size_t)(m0 + ar0) * lda + ac0;
  const _Float16* Arow1 = Ag + (size_t)(m0 + ar0 + 64) * lda + ac0;

  // B fragment pointers: column fixed per tj, advance along K
  const _Float16* bp[4];
#pragma unroll
  for (int tj = 0; tj < 4; ++tj) {
    int colg = n0 + wn * 64 + tj * 16 + r;
    if (NGUARD) colg = (colg < N) ? colg : (N - 1);
    bp[tj] = Bg + (size_t)colg * K + half * 16;
  }

  v8f acc[2][4];
#pragma unroll
  for (int ti = 0; ti < 2; ++ti)
#pragma unroll
    for (int tj = 0; tj < 4; ++tj) {
      v8f z8 = {0.f, 0.f, 0.f, 0.f, 0.f, 0.f, 0.f, 0.f};
      acc[ti][tj] = z8;
    }

  // kick off DMA of the first A tile into buffer 0
  async_b128(sA0 + so0, Arow0);
  async_b128(sA0 + so1, Arow1);
  _Float16* sc = sA0;
  _Float16* sn = sA1;

  for (int k0 = 0; k0 < K; k0 += KC) {
    wait_async0();        // this wave's DMA into sc complete
    __syncthreads();      // all waves' DMA done AND all reads of sn finished
    if (k0 + KC < K) {    // overlap next tile DMA with this tile's WMMAs
      async_b128(sn + so0, Arow0 + k0 + KC);
      async_b128(sn + so1, Arow1 + k0 + KC);
    }
    v16h bf[4];
#pragma unroll
    for (int tj = 0; tj < 4; ++tj) {
      const _Float16* p = bp[tj] + k0;
      bf[tj] = mk16(*(const h8*)p, *(const h8*)(p + 8));
    }
    v16h af[2];
#pragma unroll
    for (int ti = 0; ti < 2; ++ti) {
      const _Float16* ap = &sc[(wm * 32 + ti * 16 + r) * LDSS + half * 16];
      af[ti] = mk16(*(const h8*)ap, *(const h8*)(ap + 8));
    }
#pragma unroll
    for (int ti = 0; ti < 2; ++ti)
#pragma unroll
      for (int tj = 0; tj < 4; ++tj)
        acc[ti][tj] = __builtin_amdgcn_wmma_f32_16x16x32_f16(
            false, af[ti], false, bf[tj], (short)0, acc[ti][tj], false, false);
    _Float16* t = sc; sc = sn; sn = t;
  }

  // Epilogue. C/D layout: VGPR i -> row i + 8*(lane/16), col = lane%16.
#pragma unroll
  for (int ti = 0; ti < 2; ++ti)
#pragma unroll
    for (int tj = 0; tj < 4; ++tj) {
      const int rowb = m0 + wm * 32 + ti * 16;
      const int colg = n0 + wn * 64 + tj * 16 + r;
      if (NGUARD && colg >= N) continue;
      const float bv = bias ? bias[colg] : 0.f;
#pragma unroll
      for (int i = 0; i < 8; ++i) {
        const int rowg = rowb + i + half * 8;
        const float v = acc[ti][tj][i] + bv;
        if (MODE == 0) {
          ((_Float16*)Out)[obase + (size_t)rowg * ldo + colg] = (_Float16)v;
        } else if (MODE == 1) {
          const float ge = 0.5f * v * (1.f + erff(v * 0.70710678118654752f));
          ((_Float16*)Out)[obase + (size_t)rowg * ldo + colg] = (_Float16)ge;
        } else if (MODE == 2) {
          const int b   = rowg >> 10;
          const int t   = rowg & 1023;
          const int r0i = t >> 5, r1i = t & 31;
          const int p   = colg >> 8;
          const int p0i = p >> 3, p1i = p & 7;
          const int c   = colg & 255;
          const size_t idx =
              (((size_t)b * 256 + (r0i * 8 + p0i)) * 256 + (r1i * 8 + p1i)) * 256 + c;
          ((float*)Out)[idx] = v + resid[idx];
        } else {
          float* of = (float*)Out;
          const size_t idx = obase + (size_t)rowg * ldo + colg;
          of[idx] = of[idx] + v;
        }
      }
    }
}

// ---------------------------------------------------------------------------
// Attention scores + relative-position bias + softmax -> probs (f16).
// Block: (qtile, head, batch). 8 waves; wave w handles key tiles w, w+8, ...
// Q.K^T: K=DH=32 -> one 16x16x32 WMMA per key tile (B-fragment = rows of K).
// rpe index is affine in the accumulator row i (query base is a multiple of 8,
// so i0/i1 never wrap within the 8 rows) -> one base index per key tile.
// ---------------------------------------------------------------------------
__global__ __launch_bounds__(256) void k_attn_scores(
    const _Float16* __restrict__ qkv, const float* __restrict__ rpe,
    _Float16* __restrict__ probs) {
  __shared__ float smax[8][16];
  __shared__ float ssum[8][16];
  const int lane = threadIdx.x & 31;
  const int wave = threadIdx.x >> 5;
  const int qt = blockIdx.x;   // 0..63
  const int h  = blockIdx.y;   // 0..7
  const int b  = blockIdx.z;   // 0..1
  const int r    = lane & 15;
  const int half = lane >> 4;

  const int qrow = qt * 16 + r;
  const _Float16* qp = qkv + ((size_t)(b * LL + qrow)) * 768 + h * DH_ + half * 16;
  const v16h aq = mk16(*(const h8*)qp, *(const h8*)(qp + 8));

  v8f acc[8];
#pragma unroll
  for (int t = 0; t < 8; ++t) {
    const int key = (wave + t * 8) * 16 + r;
    const _Float16* kp =
        qkv + ((size_t)(b * LL + key)) * 768 + 256 + h * DH_ + half * 16;
    const v16h bk = mk16(*(const h8*)kp, *(const h8*)(kp + 8));
    v8f cz = {0.f, 0.f, 0.f, 0.f, 0.f, 0.f, 0.f, 0.f};
    acc[t] = __builtin_amdgcn_wmma_f32_16x16x32_f16(
        false, aq, false, bk, (short)0, cz, false, false);
  }

  const float scale = 0.17677669529663687f;  // 32^-0.5
  const int n_base = qt * 16 + half * 8;     // multiple of 8
  const int i0  = n_base >> 5;               // constant over the 8 rows
  const int i1b = n_base & 31;
#pragma unroll
  for (int t = 0; t < 8; ++t) {
    const int m  = (wave + t * 8) * 16 + r;  // key index
    const int j0 = m >> 5, j1 = m & 31;
    const int idxb = (i0 - j0 + 31) * 63 + (i1b - j1 + 31);
    const float* rp = rpe + (size_t)idxb * NHEAD + h;
#pragma unroll
    for (int i = 0; i < 8; ++i)
      acc[t][i] = acc[t][i] * scale + rp[i * NHEAD];
  }

  float rmax[8];
#pragma unroll
  for (int i = 0; i < 8; ++i) {
    float mx = acc[0][i];
#pragma unroll
    for (int t = 1; t < 8; ++t) mx = fmaxf(mx, acc[t][i]);
    rmax[i] = mx;
  }
#pragma unroll
  for (int msk = 8; msk >= 1; msk >>= 1)
#pragma unroll
    for (int i = 0; i < 8; ++i)
      rmax[i] = fmaxf(rmax[i], __shfl_xor(rmax[i], msk, 32));
  if (r == 0) {
#pragma unroll
    for (int i = 0; i < 8; ++i) smax[wave][half * 8 + i] = rmax[i];
  }
  __syncthreads();
  float gmax[8];
#pragma unroll
  for (int i = 0; i < 8; ++i) {
    float mx = -3.4e38f;
    for (int w = 0; w < 8; ++w) mx = fmaxf(mx, smax[w][half * 8 + i]);
    gmax[i] = mx;
  }
  float rsum[8] = {0.f, 0.f, 0.f, 0.f, 0.f, 0.f, 0.f, 0.f};
#pragma unroll
  for (int t = 0; t < 8; ++t)
#pragma unroll
    for (int i = 0; i < 8; ++i) {
      const float e = __expf(acc[t][i] - gmax[i]);
      acc[t][i] = e;
      rsum[i] += e;
    }
#pragma unroll
  for (int msk = 8; msk >= 1; msk >>= 1)
#pragma unroll
    for (int i = 0; i < 8; ++i) rsum[i] += __shfl_xor(rsum[i], msk, 32);
  if (r == 0) {
#pragma unroll
    for (int i = 0; i < 8; ++i) ssum[wave][half * 8 + i] = rsum[i];
  }
  __syncthreads();
  float rinv[8];
#pragma unroll
  for (int i = 0; i < 8; ++i) {
    float s = 0.f;
    for (int w = 0; w < 8; ++w) s += ssum[w][half * 8 + i];
    rinv[i] = 1.f / s;
  }
  _Float16* pb = probs + (((size_t)(b * NHEAD + h)) * LL + qt * 16) * LL;
#pragma unroll
  for (int t = 0; t < 8; ++t) {
    const int m = (wave + t * 8) * 16 + r;
#pragma unroll
    for (int i = 0; i < 8; ++i) {
      const int row = half * 8 + i;
      pb[(size_t)row * LL + m] = (_Float16)(acc[t][i] * rinv[i]);
    }
  }
}

// ---------------------------------------------------------------------------
extern "C" void kernel_launch(void* const* d_in, const int* in_sizes, int n_in,
                              void* d_out, int out_size, void* d_ws, size_t ws_size,
                              hipStream_t stream) {
  (void)in_sizes; (void)n_in; (void)out_size; (void)ws_size;
  const float* x      = (const float*)d_in[0];
  const float* ln1_g  = (const float*)d_in[1];
  const float* ln1_b  = (const float*)d_in[2];
  const float* ln2_g  = (const float*)d_in[3];
  const float* ln2_b  = (const float*)d_in[4];
  const float* rpe    = (const float*)d_in[5];
  const float* w_down = (const float*)d_in[6];
  const float* b_down = (const float*)d_in[7];
  const float* w_up   = (const float*)d_in[8];
  const float* b_up   = (const float*)d_in[9];
  const float* w_qkv  = (const float*)d_in[10];
  const float* b_qkv  = (const float*)d_in[11];
  const float* w_proj = (const float*)d_in[12];
  const float* b_proj = (const float*)d_in[13];
  const float* w_fc1  = (const float*)d_in[14];
  const float* b_fc1  = (const float*)d_in[15];
  const float* w_fc2  = (const float*)d_in[16];
  const float* b_fc2  = (const float*)d_in[17];
  float* out = (float*)d_out;

  char* ws = (char*)d_ws;
  size_t off = 0;
  auto take = [&](size_t halves) -> _Float16* {
    _Float16* p = (_Float16*)(ws + off);
    off += ((halves * 2 + 255) / 256) * 256;
    return p;
  };
  _Float16* A_down = take(33554432ull);           // (2048, 16384) LN1+unfold
  _Float16* Wd     = take(256ull * 16384);        // w_down   (N,K)
  _Float16* Wqkv   = take(768ull * 256);          // w_qkv    (N,K)
  _Float16* Wproj  = take(256ull * 256);          // w_proj   (N,K)
  _Float16* Wup    = take(16384ull * 256);        // w_up     (N,K)
  _Float16* W1     = take(1024ull * 256);         // w_fc1    (N,K)
  _Float16* W2     = take(256ull * 1024);         // w_fc2    (N,K)
  _Float16* y2     = take(2048ull * 256);         // down output
  _Float16* qkvb   = take(2048ull * 768);         // qkv output
  _Float16* probs  = take(16ull * 1024 * 1024);   // softmax(QK^T)
  _Float16* vT     = take(16ull * 32 * 1024);     // V transposed per (b,h)
  _Float16* attn_o = take(2048ull * 256);         // P.V
  _Float16* proj_o = take(2048ull * 256);         // attn projection
  _Float16* h2     = take(131072ull * 256);       // LN2 output
  _Float16* gbuf   = take(16384ull * 1024);       // GELU(fc1) chunk

  auto cg = [](size_t n) { return dim3((unsigned)((n + 1023) / 1024)); };

  // ---- weight converts (f32 -> f16, keep (N,K) layout = B^T) ----
  k_convert_f16<<<cg(256ull * 16384), 256, 0, stream>>>(w_down, Wd, 256ll * 16384);
  k_convert_f16<<<cg(768ull * 256), 256, 0, stream>>>(w_qkv, Wqkv, 768ll * 256);
  k_convert_f16<<<cg(256ull * 256), 256, 0, stream>>>(w_proj, Wproj, 256ll * 256);
  k_convert_f16<<<cg(16384ull * 256), 256, 0, stream>>>(w_up, Wup, 16384ll * 256);
  k_convert_f16<<<cg(1024ull * 256), 256, 0, stream>>>(w_fc1, W1, 1024ll * 256);
  k_convert_f16<<<cg(256ull * 1024), 256, 0, stream>>>(w_fc2, W2, 256ll * 1024);

  // ---- LN1 + unfold rearrange ----
  k_ln<<<dim3(131072 / 8), 256, 0, stream>>>(x, ln1_g, ln1_b, A_down, 1);

  // ---- down projection: (2048x16384)*(16384x256) -> y2 f16 ----
  k_gemm<0, false><<<dim3(256 / NBLK, 2048 / MBLK, 1), 256, 0, stream>>>(
      A_down, 16384, 0, 0, Wd, 0, 0, y2, 256, 0, 0,
      b_down, nullptr, 2048, 256, 16384, 1);

  // ---- qkv: (2048x256)*(256x768) -> qkvb f16 ----
  k_gemm<0, false><<<dim3(768 / NBLK, 2048 / MBLK, 1), 256, 0, stream>>>(
      y2, 256, 0, 0, Wqkv, 0, 0, qkvb, 768, 0, 0,
      b_qkv, nullptr, 2048, 768, 256, 1);

  // ---- attention scores + rpe + softmax -> probs f16; build V^T ----
  k_attn_scores<<<dim3(64, NHEAD, 2), 256, 0, stream>>>(qkvb, rpe, probs);
  k_build_vT<<<dim3(16 * 32 * 1024 / 256), 256, 0, stream>>>(qkvb, vT);

  // ---- P.V per (b,h): (1024x1024)*(1024x32) -> attn_o ----
  k_gemm<0, true><<<dim3(1, 1024 / MBLK, 16), 256, 0, stream>>>(
      probs, 1024, 8ll * LL * LL, (long)LL * LL,
      vT, 8ll * 32 * LL, 32ll * LL,
      attn_o, 256, (long)LL * 256, 32,
      nullptr, nullptr, 1024, 32, 1024, NHEAD);

  // ---- proj: (2048x256)*(256x256) -> proj_o ----
  k_gemm<0, false><<<dim3(256 / NBLK, 2048 / MBLK, 1), 256, 0, stream>>>(
      attn_o, 256, 0, 0, Wproj, 0, 0, proj_o, 256, 0, 0,
      b_proj, nullptr, 2048, 256, 256, 1);

  // ---- up projection + fold + residual -> d_out (x2, f32) ----
  k_gemm<2, false><<<dim3(16384 / NBLK, 2048 / MBLK, 1), 256, 0, stream>>>(
      proj_o, 256, 0, 0, Wup, 0, 0, out, 0, 0, 0,
      b_up, x, 2048, 16384, 256, 1);

  // ---- LN2 over x2 -> h2 f16 ----
  k_ln<<<dim3(131072 / 8), 256, 0, stream>>>(out, ln2_g, ln2_b, h2, 0);

  // ---- MLP, chunked over rows (8 x 16384) ----
  for (int c = 0; c < 8; ++c) {
    const size_t ro = (size_t)c * 16384;
    k_gemm<1, false><<<dim3(1024 / NBLK, 16384 / MBLK, 1), 256, 0, stream>>>(
        h2 + ro * 256, 256, 0, 0, W1, 0, 0, gbuf, 1024, 0, 0,
        b_fc1, nullptr, 16384, 1024, 256, 1);
    k_gemm<3, false><<<dim3(256 / NBLK, 16384 / MBLK, 1), 256, 0, stream>>>(
        gbuf, 1024, 0, 0, W2, 0, 0, out + ro * 256, 256, 0, 0,
        b_fc2, nullptr, 16384, 256, 1024, 1);
  }
}
